// MatchRNN_62964220559991
// MI455X (gfx1250) — matrix operand order repaired and checked by reading, not compiled, not run on hardware
//
#include <hip/hip_runtime.h>
#include <hip/hip_bf16.h>
#include <math.h>

// ---------------------------------------------------------------------------
// Match-LSTM (bidirectional) for MI455X / gfx1250.
// Persistent workgroup per direction; WMMA bf16 16x16x32 for all per-step
// GEMMs; LDS-staged activations; native v_tanh transcendental when available;
// global_prefetch of next-step activations; precompute kernels for
// wh/hp/flip/bf16.
// ---------------------------------------------------------------------------

typedef __attribute__((ext_vector_type(16))) __bf16 v16bf;
typedef __attribute__((ext_vector_type(8)))  __bf16 v8bf;
typedef __attribute__((ext_vector_type(8)))  float  v8f;

#define P_LEN 300
#define Q_LEN 50
#define BATCH 32
#define DDIM  256
#define HDIM  256

// LDS row strides (padded)
#define SH_H_LD  272   // bf16, h (32 x 256)
#define SH_Z_LD  528   // bf16, z (32 x 512)
#define SH_HR_LD 264   // f32,  hr / h_alpha (32 x 256)
#define SH_AL_LD 56    // f32,  alpha (32 x 50)

static constexpr size_t SMEM_H  = (size_t)32 * SH_H_LD * 2;   // 17408
static constexpr size_t SMEM_Z  = (size_t)32 * SH_Z_LD * 2;   // 33792
static constexpr size_t SMEM_HR = (size_t)32 * SH_HR_LD * 4;  // 33792
static constexpr size_t SMEM_AL = (size_t)32 * SH_AL_LD * 4;  // 7168
static constexpr size_t SMEM_TOTAL = SMEM_H + SMEM_Z + SMEM_HR + SMEM_AL;

static __device__ __forceinline__ float fast_tanh(float x) {
#if __has_builtin(__builtin_amdgcn_tanhf)
  return __builtin_amdgcn_tanhf(x);           // v_tanh_f32 (single trans op)
#elif __has_builtin(__builtin_amdgcn_tanh_f32)
  return __builtin_amdgcn_tanh_f32(x);
#else
  x = fminf(15.0f, fmaxf(-15.0f, x));
  float e = __expf(2.0f * x);
  return (e - 1.0f) / (e + 1.0f);
#endif
}
static __device__ __forceinline__ float fast_sigmoid(float x) {
  // sigmoid(x) = (1 + tanh(x/2)) / 2 : one transcendental instead of exp+div
  return 0.5f * fast_tanh(0.5f * x) + 0.5f;
}

static __device__ __forceinline__ v8f wmma_bf16(v16bf a, v16bf b, v8f c) {
  // v_wmma_f32_16x16x32_bf16
  return __builtin_amdgcn_wmma_f32_16x16x32_bf16(
      /*neg_a=*/false, a, /*neg_b=*/false, b,
      /*c_mod=*/(short)0, c, /*reuse_a=*/false, /*reuse_b=*/false);
}

// A-operand (16x32 bf16) per-lane load from an LDS row:
// lane l, row = l&15; K contiguous [kb, kb+16),
// kb = kt*32 + (l>>4)*16.
static __device__ __forceinline__ v16bf load_a_row(const __bf16* row, int kb) {
  v8bf lo = *(const v8bf*)(row + kb);
  v8bf hi = *(const v8bf*)(row + kb + 8);
  return __builtin_shufflevector(lo, hi, 0, 1, 2, 3, 4, 5, 6, 7,
                                 8, 9, 10, 11, 12, 13, 14, 15);
}

// B-operand (32x16 bf16): lane l holds column n = l&15,
// K contiguous 16 values starting at kt*32 + (l>>4)*16. W is row-major [N][K].
static __device__ __forceinline__ v16bf load_b_row(const __bf16* W, int n,
                                                   int ldk, int k0) {
  return *(const v16bf*)(W + (size_t)n * ldk + k0);
}

// ------------------------------ prep kernels -------------------------------

__global__ void lengths_kernel(const int* __restrict__ cm, int* __restrict__ len) {
  int b = threadIdx.x;
  if (b < BATCH) {
    int s = 0;
    for (int p = 0; p < P_LEN; ++p) s += cm[b * P_LEN + p];
    len[b] = s;
  }
}

__global__ void flip_kernel(const float* __restrict__ cv,
                            const int* __restrict__ len,
                            float* __restrict__ cvf) {
  size_t i = (size_t)blockIdx.x * blockDim.x + threadIdx.x;
  const size_t total = (size_t)P_LEN * BATCH * DDIM;
  if (i >= total) return;
  int d = (int)(i % DDIM);
  size_t r = i / DDIM;
  int b = (int)(r % BATCH);
  int p = (int)(r / BATCH);
  int L = len[b];
  int idx = (p < L) ? (L - 1 - p) : p;
  cvf[((size_t)p * BATCH + b) * DDIM + d] =
      cv[((size_t)idx * BATCH + b) * DDIM + d];
}

__global__ void cvt_bf16_kernel(const float* __restrict__ src,
                                __bf16* __restrict__ dst, int n) {
  int i = blockIdx.x * blockDim.x + threadIdx.x;
  if (i < n) dst[i] = (__bf16)src[i];
}

// Y[r][h] = dot(X[r,:256], W[h,:256]) + bias[h]
__global__ void proj_kernel(const float* __restrict__ X,
                            const float* __restrict__ W,
                            const float* __restrict__ bias,
                            float* __restrict__ Y, int rows) {
  int i = blockIdx.x * blockDim.x + threadIdx.x;
  if (i >= rows * HDIM) return;
  int h = i & (HDIM - 1);
  int r = i >> 8;
  const float4* x = (const float4*)(X + (size_t)r * DDIM);
  const float4* w = (const float4*)(W + (size_t)h * DDIM);
  float s = 0.0f;
#pragma unroll 8
  for (int k = 0; k < DDIM / 4; ++k) {
    float4 a = x[k], b = w[k];
    s += a.x * b.x + a.y * b.y + a.z * b.z + a.w * b.w;
  }
  Y[i] = s + bias[h];
}

// ------------------------------ main scan ----------------------------------

__global__ __launch_bounds__(512) void match_scan_kernel(
    const float* __restrict__ cv0,      // (P,B,D) original
    const float* __restrict__ cv1,      // (P,B,D) masked-flipped
    const float* __restrict__ qv,       // (Q,B,D)
    const int*   __restrict__ qmask,    // (B,Q)
    const float* __restrict__ wh_ws,    // [2](Q,B,H)  qv@Wq^T+bq
    const float* __restrict__ hp_ws,    // [2](P,B,H)  cv_dir@Wp^T+bp
    const __bf16* __restrict__ Wr_bf,   // [2](H,H)
    const __bf16* __restrict__ Wih_bf,  // [2](4H,2D)
    const __bf16* __restrict__ Whh_bf,  // [2](4H,H)
    const float* __restrict__ br,       // (2,H)
    const float* __restrict__ Wg,       // (2,H)
    const float* __restrict__ bg,       // (2,)
    const float* __restrict__ ln_g,     // (2,2D)
    const float* __restrict__ ln_b,     // (2,2D)
    const float* __restrict__ b_ih,     // (2,4H)
    const float* __restrict__ b_hh,     // (2,4H)
    float* __restrict__ out,            // (P,B,2H) : dir0 writes [:,:,:256]
    float* __restrict__ left_ws)        // (P,B,H)  : dir1 writes here
{
  const int dir = blockIdx.x;
  const int tid = threadIdx.x;
  const int wv = tid >> 5;       // wave 0..15
  const int ln = tid & 31;       // lane within wave
  const int lhalf = ln >> 4;     // 0/1 (lane half)
  const int l15 = ln & 15;

  extern __shared__ char smem[];
  __bf16* sh_h = (__bf16*)smem;                               // 32 x SH_H_LD
  __bf16* sh_z = (__bf16*)(smem + SMEM_H);                    // 32 x SH_Z_LD
  float* sh_hr = (float*)(smem + SMEM_H + SMEM_Z);            // 32 x SH_HR_LD
  float* sh_ha = sh_hr;                                       // overlay (hr dead)
  float* sh_al = (float*)(smem + SMEM_H + SMEM_Z + SMEM_HR);  // 32 x SH_AL_LD

  __shared__ float s_Wg[HDIM];
  __shared__ float s_br[HDIM];
  __shared__ float s_bias[4 * HDIM];
  __shared__ float s_lng[2 * DDIM];
  __shared__ float s_lnb[2 * DDIM];
  __shared__ float s_ps[32][16];
  __shared__ float s_pq[32][16];
  __shared__ float s_stats[32][2];

  // small-parameter staging
  for (int i = tid; i < HDIM; i += 512) {
    s_Wg[i] = Wg[dir * HDIM + i];
    s_br[i] = br[dir * HDIM + i];
  }
  for (int i = tid; i < 4 * HDIM; i += 512)
    s_bias[i] = b_ih[dir * 4 * HDIM + i] + b_hh[dir * 4 * HDIM + i];
  for (int i = tid; i < 2 * DDIM; i += 512) {
    s_lng[i] = ln_g[dir * 2 * DDIM + i];
    s_lnb[i] = ln_b[dir * 2 * DDIM + i];
  }
  for (int i = tid; i < 32 * SH_H_LD; i += 512) sh_h[i] = (__bf16)0.0f;
  const float bg_s = bg[dir];

  const float* cvD  = dir ? cv1 : cv0;
  const float* whD  = wh_ws + (size_t)dir * Q_LEN * BATCH * HDIM;
  const float* hpD  = hp_ws + (size_t)dir * P_LEN * BATCH * HDIM;
  const __bf16* WrD  = Wr_bf  + (size_t)dir * HDIM * HDIM;
  const __bf16* WihD = Wih_bf + (size_t)dir * 4 * HDIM * 2 * DDIM;
  const __bf16* WhhD = Whh_bf + (size_t)dir * 4 * HDIM * HDIM;

  float cst[2][8];   // per-lane cell state for this wave's 2 gate tiles
  float hreg[2][8];
#pragma unroll
  for (int j = 0; j < 2; ++j)
#pragma unroll
    for (int r = 0; r < 8; ++r) cst[j][r] = 0.0f;

  __syncthreads();

  for (int t = 0; t < P_LEN; ++t) {
    // -------- prefetch next step's hp_t and cv_t rows into cache --------
    if (t + 1 < P_LEN) {
      // 32KB each = 256 x 128B cachelines; one line per low thread.
      if (tid < 256) {
        const float* nhp = hpD + (size_t)(t + 1) * BATCH * HDIM;
        __builtin_prefetch(nhp + tid * 32, 0, 1);   // global_prefetch_b8
        const float* ncv = cvD + (size_t)(t + 1) * BATCH * DDIM;
        __builtin_prefetch(ncv + tid * 32, 0, 1);
      }
    }

    // ---------------- Phase A: hr = h @ Wr^T + br (WMMA) ----------------
    // Wave-uniform literal trip count; unroll(1) keeps one tile job's
    // registers live at a time (no spills).
#pragma unroll 1
    for (int ji = 0; ji < 2; ++ji) {
      const int j = wv + ji * 16;
      const int mtile = j & 1;
      const int nt = j >> 1;
      const int n = nt * 16 + l15;
      const __bf16* arow = sh_h + (mtile * 16 + l15) * SH_H_LD;
      v8f acc = {};
#pragma unroll
      for (int kt = 0; kt < 8; ++kt) {
        v16bf a = load_a_row(arow, kt * 32 + lhalf * 16);
        v16bf b = load_b_row(WrD, n, HDIM, kt * 32 + lhalf * 16);
        acc = wmma_bf16(a, b, acc);
      }
      const float bias = s_br[n];
#pragma unroll
      for (int r = 0; r < 8; ++r) {
        int row = mtile * 16 + r + 8 * lhalf;
        sh_hr[row * SH_HR_LD + n] = acc[r] + bias;
      }
    }
    __syncthreads();

    // ---------------- Phase B1: attention scores ----------------
    const float* hp_t = hpD + (size_t)t * BATCH * HDIM;
    for (int pair = tid; pair < Q_LEN * BATCH; pair += 512) {
      const int q = pair >> 5, b = pair & 31;
      const float4* wh4 = (const float4*)(whD + ((size_t)q * BATCH + b) * HDIM);
      const float4* hp4 = (const float4*)(hp_t + (size_t)b * HDIM);
      const float4* hr4 = (const float4*)(sh_hr + b * SH_HR_LD);
      const float4* wg4 = (const float4*)s_Wg;
      float s = 0.0f;
#pragma unroll 2
      for (int k = 0; k < HDIM / 4; ++k) {
        float4 a = wh4[k], h = hp4[k], c = hr4[k], g = wg4[k];
        s += g.x * fast_tanh(a.x + h.x + c.x);
        s += g.y * fast_tanh(a.y + h.y + c.y);
        s += g.z * fast_tanh(a.z + h.z + c.z);
        s += g.w * fast_tanh(a.w + h.w + c.w);
      }
      sh_al[b * SH_AL_LD + q] = s + bg_s;
    }
    __syncthreads();

    // ---------------- Phase B2: masked softmax over Q ----------------
    if (tid < 32) {
      const int b = tid;
      const int* qm = qmask + b * Q_LEN;
      float mx = -1e30f;
      for (int q = 0; q < Q_LEN; ++q)
        if (qm[q]) mx = fmaxf(mx, sh_al[b * SH_AL_LD + q]);
      float sum = 0.0f;
      for (int q = 0; q < Q_LEN; ++q) {
        float e = qm[q] ? __expf(sh_al[b * SH_AL_LD + q] - mx) : 0.0f;
        sh_al[b * SH_AL_LD + q] = e;
        sum += e;
      }
      float inv = 1.0f / sum;
      for (int q = 0; q < Q_LEN; ++q) sh_al[b * SH_AL_LD + q] *= inv;
    }
    __syncthreads();

    // ---------------- Phase B3: h_alpha = alpha @ qv ----------------
    {
      const int b = tid >> 4;
      const int d0 = (tid & 15) * 16;
      float accv[16];
#pragma unroll
      for (int i = 0; i < 16; ++i) accv[i] = 0.0f;
      for (int q = 0; q < Q_LEN; ++q) {
        const float a = sh_al[b * SH_AL_LD + q];
        const float* qr = qv + ((size_t)q * BATCH + b) * DDIM + d0;
#pragma unroll
        for (int i = 0; i < 16; ++i) accv[i] += a * qr[i];
      }
#pragma unroll
      for (int i = 0; i < 16; ++i) sh_ha[b * SH_HR_LD + d0 + i] = accv[i];
    }
    __syncthreads();

    // ---------------- Phase B4: layernorm(z) -> sh_z (bf16) ----------------
    const float* cvt = cvD + (size_t)t * BATCH * DDIM;
    {
      const int b = tid >> 4, seg = tid & 15;
      const int k0 = seg * 32;
      float s = 0.0f, ss = 0.0f;
#pragma unroll 4
      for (int i = 0; i < 32; ++i) {
        int k = k0 + i;
        float x = (k < DDIM) ? cvt[(size_t)b * DDIM + k]
                             : sh_ha[b * SH_HR_LD + (k - DDIM)];
        s += x;
        ss += x * x;
      }
      s_ps[b][seg] = s;
      s_pq[b][seg] = ss;
    }
    __syncthreads();
    if (tid < 32) {
      const int b = tid;
      float s = 0.0f, ss = 0.0f;
      for (int i = 0; i < 16; ++i) { s += s_ps[b][i]; ss += s_pq[b][i]; }
      float mu = s * (1.0f / 512.0f);
      float var = ss * (1.0f / 512.0f) - mu * mu;
      s_stats[b][0] = mu;
      s_stats[b][1] = rsqrtf(var + 1e-5f);
    }
    __syncthreads();
    {
      const int b = tid >> 4, seg = tid & 15;
      const int k0 = seg * 32;
      const float mu = s_stats[b][0], rs = s_stats[b][1];
#pragma unroll 4
      for (int i = 0; i < 32; ++i) {
        int k = k0 + i;
        float x = (k < DDIM) ? cvt[(size_t)b * DDIM + k]
                             : sh_ha[b * SH_HR_LD + (k - DDIM)];
        float zn = (x - mu) * rs * s_lng[k] + s_lnb[k];
        sh_z[b * SH_Z_LD + k] = (__bf16)zn;
      }
    }
    __syncthreads();

    // ---------------- Phase C: gates GEMM (WMMA) + LSTM cell ----------------
#pragma unroll 1
    for (int ji = 0; ji < 2; ++ji) {
      const int j = wv + ji * 16;
      const int mtile = j & 1;
      const int nt = j >> 1;
      const int ncol = nt * 16 + l15;
      const __bf16* arz = sh_z + (mtile * 16 + l15) * SH_Z_LD;
      const __bf16* arh = sh_h + (mtile * 16 + l15) * SH_H_LD;
      v8f ai = {}, af = {}, ag = {}, ao = {};
      // K over z (512)
#pragma unroll 2
      for (int kt = 0; kt < 16; ++kt) {
        v16bf a = load_a_row(arz, kt * 32 + lhalf * 16);
        const int k0 = kt * 32 + lhalf * 16;
        v16bf b0 = load_b_row(WihD, ncol,       2 * DDIM, k0);
        v16bf b1 = load_b_row(WihD, ncol + 256, 2 * DDIM, k0);
        v16bf b2 = load_b_row(WihD, ncol + 512, 2 * DDIM, k0);
        v16bf b3 = load_b_row(WihD, ncol + 768, 2 * DDIM, k0);
        ai = wmma_bf16(a, b0, ai);
        af = wmma_bf16(a, b1, af);
        ag = wmma_bf16(a, b2, ag);
        ao = wmma_bf16(a, b3, ao);
      }
      // K over h (256)
#pragma unroll 2
      for (int kt = 0; kt < 8; ++kt) {
        v16bf a = load_a_row(arh, kt * 32 + lhalf * 16);
        const int k0 = kt * 32 + lhalf * 16;
        v16bf b0 = load_b_row(WhhD, ncol,       HDIM, k0);
        v16bf b1 = load_b_row(WhhD, ncol + 256, HDIM, k0);
        v16bf b2 = load_b_row(WhhD, ncol + 512, HDIM, k0);
        v16bf b3 = load_b_row(WhhD, ncol + 768, HDIM, k0);
        ai = wmma_bf16(a, b0, ai);
        af = wmma_bf16(a, b1, af);
        ag = wmma_bf16(a, b2, ag);
        ao = wmma_bf16(a, b3, ao);
      }
      const float bi = s_bias[ncol];
      const float bf = s_bias[ncol + 256];
      const float bgt = s_bias[ncol + 512];
      const float bo = s_bias[ncol + 768];
#pragma unroll
      for (int r = 0; r < 8; ++r) {
        float iv = fast_sigmoid(ai[r] + bi);
        float fv = fast_sigmoid(af[r] + bf);
        float gv = fast_tanh(ag[r] + bgt);
        float ov = fast_sigmoid(ao[r] + bo);
        float cn = fv * cst[ji][r] + iv * gv;
        cst[ji][r] = cn;
        hreg[ji][r] = ov * fast_tanh(cn);
      }
    }
    __syncthreads();  // all reads of sh_h (h_{t-1}) finished

    // publish h_t and write outputs
#pragma unroll 1
    for (int ji = 0; ji < 2; ++ji) {
      const int j = wv + ji * 16;
      const int mtile = j & 1;
      const int nt = j >> 1;
      const int ncol = nt * 16 + l15;
#pragma unroll
      for (int r = 0; r < 8; ++r) {
        const int b = mtile * 16 + r + 8 * lhalf;
        const float hn = hreg[ji][r];
        sh_h[b * SH_H_LD + ncol] = (__bf16)hn;
        if (dir == 0)
          out[((size_t)t * BATCH + b) * (2 * HDIM) + ncol] = hn;
        else
          left_ws[((size_t)t * BATCH + b) * HDIM + ncol] = hn;
      }
    }
    __syncthreads();
  }
}

// --------------------------- finalize: mask + flip --------------------------

__global__ void finalize_kernel(const int* __restrict__ cm,
                                const int* __restrict__ len,
                                const float* __restrict__ left_ws,
                                float* __restrict__ out) {
  size_t i = (size_t)blockIdx.x * blockDim.x + threadIdx.x;
  const size_t total = (size_t)P_LEN * BATCH * HDIM;
  if (i >= total) return;
  int h = (int)(i & (HDIM - 1));
  size_t r = i >> 8;
  int b = (int)(r % BATCH);
  int p = (int)(r / BATCH);
  float m = (float)cm[b * P_LEN + p];
  int L = len[b];
  int idx = (p < L) ? (L - 1 - p) : p;
  size_t o = ((size_t)p * BATCH + b) * (2 * HDIM);
  out[o + h] *= m;
  out[o + HDIM + h] = m * left_ws[((size_t)idx * BATCH + b) * HDIM + h];
}

// ------------------------------ host launcher -------------------------------

extern "C" void kernel_launch(void* const* d_in, const int* in_sizes, int n_in,
                              void* d_out, int out_size, void* d_ws,
                              size_t ws_size, hipStream_t stream) {
  const float* cv   = (const float*)d_in[0];   // (P,B,D)
  const int*   cm   = (const int*)d_in[1];     // (B,P)
  const float* qv   = (const float*)d_in[2];   // (Q,B,D)
  const int*   qm   = (const int*)d_in[3];     // (B,Q)
  const float* Wq   = (const float*)d_in[4];   // (2,H,D)
  const float* bq   = (const float*)d_in[5];   // (2,H)
  const float* Wp   = (const float*)d_in[6];   // (2,H,D)
  const float* bp   = (const float*)d_in[7];   // (2,H)
  const float* Wr   = (const float*)d_in[8];   // (2,H,H)
  const float* br   = (const float*)d_in[9];   // (2,H)
  const float* Wg   = (const float*)d_in[10];  // (2,H)
  const float* bg   = (const float*)d_in[11];  // (2,)
  const float* lng  = (const float*)d_in[12];  // (2,2D)
  const float* lnb  = (const float*)d_in[13];  // (2,2D)
  const float* Wih  = (const float*)d_in[14];  // (2,4H,2D)
  const float* Whh  = (const float*)d_in[15];  // (2,4H,H)
  const float* bih  = (const float*)d_in[16];  // (2,4H)
  const float* bhh  = (const float*)d_in[17];  // (2,4H)
  float* out = (float*)d_out;                  // (P,B,2H)

  // workspace carve (256B aligned)
  char* w = (char*)d_ws;
  auto alloc = [&](size_t bytes) -> char* {
    char* p = w;
    w += (bytes + 255) & ~(size_t)255;
    return p;
  };
  int*    len_ws  = (int*)alloc(BATCH * sizeof(int));
  float*  cv_flip = (float*)alloc(sizeof(float) * P_LEN * BATCH * DDIM);
  float*  wh_ws   = (float*)alloc(sizeof(float) * 2 * Q_LEN * BATCH * HDIM);
  float*  hp_ws   = (float*)alloc(sizeof(float) * 2 * P_LEN * BATCH * HDIM);
  __bf16* Wr_bf   = (__bf16*)alloc(sizeof(__bf16) * 2 * HDIM * HDIM);
  __bf16* Wih_bf  = (__bf16*)alloc(sizeof(__bf16) * 2 * 4 * HDIM * 2 * DDIM);
  __bf16* Whh_bf  = (__bf16*)alloc(sizeof(__bf16) * 2 * 4 * HDIM * HDIM);
  float*  left_ws = (float*)alloc(sizeof(float) * P_LEN * BATCH * HDIM);

  // 1) sequence lengths
  lengths_kernel<<<1, 32, 0, stream>>>(cm, len_ws);

  // 2) masked flip of content for the left direction
  {
    size_t n = (size_t)P_LEN * BATCH * DDIM;
    flip_kernel<<<(int)((n + 255) / 256), 256, 0, stream>>>(cv, len_ws, cv_flip);
  }

  // 3) bf16 weight conversion (both directions packed)
  {
    int n1 = 2 * HDIM * HDIM;
    cvt_bf16_kernel<<<(n1 + 255) / 256, 256, 0, stream>>>(Wr, Wr_bf, n1);
    int n2 = 2 * 4 * HDIM * 2 * DDIM;
    cvt_bf16_kernel<<<(n2 + 255) / 256, 256, 0, stream>>>(Wih, Wih_bf, n2);
    int n3 = 2 * 4 * HDIM * HDIM;
    cvt_bf16_kernel<<<(n3 + 255) / 256, 256, 0, stream>>>(Whh, Whh_bf, n3);
  }

  // 4) precompute wh = qv@Wq^T+bq and hp = cv_dir@Wp^T+bp, per direction
  for (int dir = 0; dir < 2; ++dir) {
    {
      int rows = Q_LEN * BATCH;
      int n = rows * HDIM;
      proj_kernel<<<(n + 255) / 256, 256, 0, stream>>>(
          qv, Wq + (size_t)dir * HDIM * DDIM, bq + dir * HDIM,
          wh_ws + (size_t)dir * Q_LEN * BATCH * HDIM, rows);
    }
    {
      int rows = P_LEN * BATCH;
      int n = rows * HDIM;
      proj_kernel<<<(n + 255) / 256, 256, 0, stream>>>(
          dir ? cv_flip : cv, Wp + (size_t)dir * HDIM * DDIM, bp + dir * HDIM,
          hp_ws + (size_t)dir * P_LEN * BATCH * HDIM, rows);
    }
  }

  // 5) persistent recurrent scan: one workgroup per direction
  match_scan_kernel<<<2, 512, SMEM_TOTAL, stream>>>(
      cv, cv_flip, qv, qm, wh_ws, hp_ws, Wr_bf, Wih_bf, Whh_bf, br, Wg, bg,
      lng, lnb, bih, bhh, out, left_ws);

  // 6) apply content mask + un-flip left outputs into out[:,:,256:]
  {
    size_t n = (size_t)P_LEN * BATCH * HDIM;
    finalize_kernel<<<(int)((n + 255) / 256), 256, 0, stream>>>(cm, len_ws,
                                                               left_ws, out);
  }
}
